// KANGNN_6940667150995
// MI455X (gfx1250) — matrix operand order repaired and checked by reading, not compiled, not run on hardware
//
#include <hip/hip_runtime.h>
#include <hip/hip_bf16.h>
#include <math.h>

typedef _Float16 v16h __attribute__((ext_vector_type(16)));
typedef float    v8f  __attribute__((ext_vector_type(8)));

#define NBF 11  // number of b-spline basis funcs

// epilogue modes for the fused WMMA linear kernel
#define EPI_RAW    0   // Y = X@W + bias
#define EPI_SPLINE 1   // Y = kan_act(X@W + bias)
#define EPI_MSG    2   // Y = kan_act(X@W); aux = dis[row]^2 * Y   (per-channel spline)
#define EPI_LNRES  3   // aux(=h) += layernorm(X@W)*g + b         (ncols must be 32)

// ---------------- spline activation: ONE out-of-line copy shared by all call sites ----------
__device__ __noinline__ float kan_act(float x, const float* __restrict__ coef, float blend) {
    // uniform grid: 14 knots on [-3,3], step = 6/13. Cox-de Boor, order 3 (degree 2).
    const float lo = -3.0f;
    const float step = 6.0f / 13.0f;
    float xc = fminf(fmaxf(x, -3.0f), 3.0f);
    float b[13];
#pragma unroll
    for (int j = 0; j < 13; ++j) {
        float gj  = lo + step * j;
        float gj1 = lo + step * (j + 1);
        b[j] = (xc >= gj && xc < gj1) ? 1.0f : 0.0f;
    }
#pragma unroll
    for (int k = 1; k <= 2; ++k) {
        float inv = 1.0f / (step * k);  // ld == rd == k*step (uniform grid)
#pragma unroll
        for (int j = 0; j < 13 - 1; ++j) {
            if (j < 13 - k) {
                float gj = lo + step * j;
                float left  = (xc - gj) * inv;
                float right = ((gj + step * (k + 1)) - xc) * inv;
                b[j] = left * b[j] + right * b[j + 1];
            }
        }
    }
    float s = 0.0f;
#pragma unroll
    for (int j = 0; j < NBF; ++j) s += b[j] * coef[j];
    float sig = 1.0f / (1.0f + expf(-x));
    return x * sig + fabsf(blend) * s;
}

// ---------------- fused WMMA linear: Y[n,ncols] = epi(X[n,K] @ W[K,ncols]) ----------------
// 128 threads = 4 waves, each wave computes a 16-row tile; weights live in LDS as f16,
// transposed [ncols][Kpad] so each lane's B fragment is two contiguous 8-half runs (2x ds_load_b128).
// A fragment per lane = two contiguous 8-float runs of the row -> global_load_b128 pairs,
// row index clamped for reads. STORES ARE UNGUARDED: output buffers must have
// ceil(n_rows/64)*64 rows of capacity (workspace is padded; pad rows never consumed).
// Each lane owns a fixed output channel (col0 [, col0+16]) -> per-channel spline coef is
// a single 11-float load; layernorm = 4x shfl_xor within each 16-lane half.
// Requires K in {16,32,64}, ncols in {16,32} (EPI_LNRES: ncols==32).
__global__ __launch_bounds__(128)
void k_wmma_linear(const float* __restrict__ X, int n_rows, int K,
                   const float* __restrict__ W, int ncols,
                   float* __restrict__ Y, int mode,
                   const float* __restrict__ bias,      // EPI_RAW/SPLINE (may be null)
                   const float* __restrict__ coef,      // EPI_SPLINE/MSG
                   const float* __restrict__ blend,     // EPI_SPLINE/MSG
                   int per_channel,                     // EPI_SPLINE
                   const float* __restrict__ dis,       // EPI_MSG
                   float* __restrict__ aux,             // EPI_MSG: agg ; EPI_LNRES: h
                   const float* __restrict__ gvec,      // EPI_LNRES
                   const float* __restrict__ bvec) {    // EPI_LNRES
    __shared__ _Float16 Wl[64 * 32];              // up to Kpad=64, ncols=32 (4KB)
    const int Kpad = (K + 31) & ~31;
    const int tot  = Kpad * ncols;
    for (int i = threadIdx.x; i < tot; i += blockDim.x) {
        int c = i / Kpad, k = i % Kpad;
        Wl[c * Kpad + k] = (k < K) ? (_Float16)W[k * ncols + c] : (_Float16)0.0f;
    }
    __syncthreads();

    const int lane = threadIdx.x & 31;
    const int wave = threadIdx.x >> 5;
    const int row0 = blockIdx.x * 64 + wave * 16;
    const int arow = row0 + (lane & 15);          // A-matrix row this lane supplies
    const int rowc = (arow < n_rows) ? arow : (n_rows - 1);  // clamped read row
    const int col0 = lane & 15;                   // B/D column (within 16-wide tile)
    const int klo_a = (lane & 16) ? 8 : 0;        // A K-layout split (ISA 7.12.2)
    const int kb_b  = (lane & 16) ? 8 : 0;        // B K-layout split
    const int nnt   = ncols >> 4;                 // 1 or 2 N-tiles
    const float* xr = X + rowc * K;

    v8f acc0 = {}; v8f acc1 = {};
    for (int ks = 0; ks < Kpad; ks += 32) {
        v16h a;
        {   // run 1: kk = klo_a + j (j=0..7) -> a[0..7]; always in-range
            float4 f0 = *(const float4*)(xr + ks + klo_a);
            float4 f1 = *(const float4*)(xr + ks + klo_a + 4);
            a[0] = (_Float16)f0.x; a[1] = (_Float16)f0.y;
            a[2] = (_Float16)f0.z; a[3] = (_Float16)f0.w;
            a[4] = (_Float16)f1.x; a[5] = (_Float16)f1.y;
            a[6] = (_Float16)f1.z; a[7] = (_Float16)f1.w;
        }
        if (ks + 16 < K) {  // run 2: kk = 16 + klo_a + j -> a[8..15]; wave-uniform validity
            float4 f0 = *(const float4*)(xr + ks + 16 + klo_a);
            float4 f1 = *(const float4*)(xr + ks + 16 + klo_a + 4);
            a[8]  = (_Float16)f0.x; a[9]  = (_Float16)f0.y;
            a[10] = (_Float16)f0.z; a[11] = (_Float16)f0.w;
            a[12] = (_Float16)f1.x; a[13] = (_Float16)f1.y;
            a[14] = (_Float16)f1.z; a[15] = (_Float16)f1.w;
        } else {
#pragma unroll
            for (int i = 8; i < 16; ++i) a[i] = (_Float16)0.0f;
        }
        {
            v16h bm;
            const _Float16* wp = &Wl[col0 * Kpad + ks + kb_b];
#pragma unroll
            for (int i = 0; i < 16; ++i) bm[i] = wp[(i < 8) ? i : i + 8];
            acc0 = __builtin_amdgcn_wmma_f32_16x16x32_f16(false, a, false, bm,
                                                          (short)0, acc0, false, false);
        }
        if (nnt > 1) {
            v16h bm;
            const _Float16* wp = &Wl[(col0 + 16) * Kpad + ks + kb_b];
#pragma unroll
            for (int i = 0; i < 16; ++i) bm[i] = wp[(i < 8) ? i : i + 8];
            acc1 = __builtin_amdgcn_wmma_f32_16x16x32_f16(false, a, false, bm,
                                                          (short)0, acc1, false, false);
        }
    }

    // D layout: VGPR r -> M = r + 8*(lane>=16); N = lane&15.  Unguarded stores (padded bufs).
    const int mbase = row0 + ((lane & 16) ? 8 : 0);

    if (mode == EPI_LNRES) {
        // layernorm over the 32 channels of each row + residual into aux(=h)
        float g0 = gvec[col0],      g1 = gvec[col0 + 16];
        float v0 = bvec[col0],      v1 = bvec[col0 + 16];
#pragma unroll
        for (int r = 0; r < 8; ++r) {
            float o0 = acc0[r], o1 = acc1[r];
            float s = o0 + o1;
            float q = o0 * o0 + o1 * o1;
#pragma unroll
            for (int m2 = 1; m2 <= 8; m2 <<= 1) {   // reduce across the 16-lane half
                s += __shfl_xor(s, m2, 32);
                q += __shfl_xor(q, m2, 32);
            }
            float mu  = s * (1.0f / 32.0f);
            float var = q * (1.0f / 32.0f) - mu * mu;
            float inv = rsqrtf(var + 1e-5f);
            int idx = (mbase + r) * 32 + col0;
            aux[idx]      += (o0 - mu) * inv * g0 + v0;
            aux[idx + 16] += (o1 - mu) * inv * g1 + v1;
        }
        return;
    }

    // common path: optional bias, optional spline (out-of-line call), optional agg-init
    {
        float bia = bias ? bias[col0] : 0.0f;
        const float* cf = coef + ((mode != EPI_RAW && per_channel) ? col0 * NBF : 0);
        float bl = (mode != EPI_RAW) ? blend[per_channel ? col0 : 0] : 0.0f;
        int idx = mbase * ncols + col0;
#pragma unroll
        for (int r = 0; r < 8; ++r) {
            float v = acc0[r] + bia;
            if (mode != EPI_RAW) v = kan_act(v, cf, bl);
            Y[idx] = v;
            if (mode == EPI_MSG) {
                float di = dis[mbase + r];
                aux[idx] = di * di * v;
            }
            idx += ncols;
        }
    }
    if (nnt > 1) {
        int c1 = col0 + 16;
        float bia = bias ? bias[c1] : 0.0f;
        const float* cf = coef + ((mode != EPI_RAW && per_channel) ? c1 * NBF : 0);
        float bl = (mode != EPI_RAW) ? blend[per_channel ? c1 : 0] : 0.0f;
        int idx = mbase * ncols + c1;
#pragma unroll
        for (int r = 0; r < 8; ++r) {
            float v = acc1[r] + bia;
            if (mode != EPI_RAW) v = kan_act(v, cf, bl);
            Y[idx] = v;
            if (mode == EPI_MSG) {
                float di = dis[mbase + r];
                aux[idx] = di * di * v;
            }
            idx += ncols;
        }
    }
}

// ---------------- elementwise kernels ----------------
__global__ void k_zero(float* __restrict__ p, int n) {
    int t = blockIdx.x * blockDim.x + threadIdx.x;
    if (t < n) p[t] = 0.0f;
}

__global__ void k_deg_edges(const int* __restrict__ ei, float* __restrict__ deg, int E) {
    int e = blockIdx.x * blockDim.x + threadIdx.x;
    if (e < E) atomicAdd(&deg[ei[E + e]], 1.0f);
}

__global__ void k_dis(const float* __restrict__ deg, float* __restrict__ dis, int n) {
    int i = blockIdx.x * blockDim.x + threadIdx.x;
    if (i < n) dis[i] = rsqrtf(deg[i] + 1.0f);   // +1 = self loop; always > 0
}

// Y[t] = kan_act(Y[t] + add[t]) ; per-channel coef/blend, C = 1<<cbits
__global__ void k_spline_add(float* __restrict__ Y, const float* __restrict__ add,
                             int n_elems, int cbits,
                             const float* __restrict__ coef, const float* __restrict__ blend) {
    int t = blockIdx.x * blockDim.x + threadIdx.x;
    if (t >= n_elems) return;
    int c = t & ((1 << cbits) - 1);
    float x = Y[t] + add[t];
    Y[t] = kan_act(x, coef + c * NBF, blend[c]);
}

// scatter: agg[col] += dis[row]*dis[col] * m[row]  (8 threads/edge, float4 each)
__global__ void k_edge_scatter(const int* __restrict__ ei, int E,
                               const float* __restrict__ dis,
                               const float* __restrict__ m,
                               float* __restrict__ agg) {
    int t = blockIdx.x * blockDim.x + threadIdx.x;
    int tot = E * 8;
    if (t >= tot) return;
    int e = t >> 3;
    int q = t & 7;
    int r = ei[e];
    int c = ei[E + e];
    float en = dis[r] * dis[c];
    float4 mv = *(const float4*)(m + r * 32 + q * 4);
    float* dst = agg + c * 32 + q * 4;
    atomicAdd(dst + 0, en * mv.x);
    atomicAdd(dst + 1, en * mv.y);
    atomicAdd(dst + 2, en * mv.z);
    atomicAdd(dst + 3, en * mv.w);
}

// ---- pooling: ordered-uint encoding for float atomic max ----
__device__ __forceinline__ unsigned enc_f(float f) {
    unsigned u = __float_as_uint(f);
    return (u & 0x80000000u) ? ~u : (u | 0x80000000u);
}
__device__ __forceinline__ float dec_f(unsigned u) {
    unsigned b = (u & 0x80000000u) ? (u & 0x7FFFFFFFu) : ~u;
    return __uint_as_float(b);
}

__global__ void k_pool_accum(const float* __restrict__ h, const int* __restrict__ batch,
                             float* __restrict__ hsum, unsigned* __restrict__ umax,
                             float* __restrict__ counts, int n_elems, int cbits) {
    int t = blockIdx.x * blockDim.x + threadIdx.x;
    if (t >= n_elems) return;
    int i = t >> cbits, c = t & ((1 << cbits) - 1);
    int g = batch[i];
    float v = h[t];
    atomicAdd(&hsum[(g << cbits) + c], v);
    atomicMax(&umax[(g << cbits) + c], enc_f(v));
    if (c == 0) atomicAdd(&counts[g], 1.0f);
}

__global__ void k_pool_final(const float* __restrict__ hsum, const unsigned* __restrict__ umax,
                             const float* __restrict__ counts, float* __restrict__ hg,
                             int n_elems, int cbits) {
    int t = blockIdx.x * blockDim.x + threadIdx.x;
    if (t >= n_elems) return;
    int C = 1 << cbits;
    int g = t >> cbits, c = t & (C - 1);
    float cnt = counts[g];
    hg[g * 2 * C + c]     = hsum[t] / fmaxf(cnt, 1.0f);
    hg[g * 2 * C + C + c] = (cnt > 0.0f) ? dec_f(umax[t]) : 0.0f;
}

__global__ void k_final(const float* __restrict__ r2, const float* __restrict__ W3,
                        const float* __restrict__ b3, float* __restrict__ out,
                        int ng) {
    const int K = 16;
    int g = blockIdx.x * blockDim.x + threadIdx.x;
    if (g >= ng) return;
    float acc = b3[0];
#pragma unroll
    for (int j = 0; j < K; ++j) acc += r2[g * K + j] * W3[j];
    out[g] = 1.0f / (1.0f + expf(-acc));
}

// ---------------- host launch ----------------
static inline int cdiv(long a, int b) { return (int)((a + b - 1) / b); }

extern "C" void kernel_launch(void* const* d_in, const int* in_sizes, int n_in,
                              void* d_out, int out_size, void* d_ws, size_t ws_size,
                              hipStream_t stream) {
    const float* x        = (const float*)d_in[0];
    const float* W_in     = (const float*)d_in[1];
    const float* b_in     = (const float*)d_in[2];
    const float* coef_in  = (const float*)d_in[3];
    const float* blend_in = (const float*)d_in[4];
    const float* Wmsg     = (const float*)d_in[5];
    const float* Wself    = (const float*)d_in[6];
    const float* Wout     = (const float*)d_in[7];
    const float* coef_msg = (const float*)d_in[8];
    const float* blend_msg= (const float*)d_in[9];
    const float* coef_upd = (const float*)d_in[10];
    const float* blend_upd= (const float*)d_in[11];
    const float* ln_g     = (const float*)d_in[12];
    const float* ln_b     = (const float*)d_in[13];
    const float* W1       = (const float*)d_in[14];
    const float* b1       = (const float*)d_in[15];
    const float* coef_r1  = (const float*)d_in[16];
    const float* blend_r1 = (const float*)d_in[17];
    const float* W2       = (const float*)d_in[18];
    const float* b2       = (const float*)d_in[19];
    const float* coef_r2  = (const float*)d_in[20];
    const float* blend_r2 = (const float*)d_in[21];
    const float* W3       = (const float*)d_in[22];
    const float* b3       = (const float*)d_in[23];
    const int*   ei       = (const int*)d_in[24];
    const int*   batch    = (const int*)d_in[25];

    const int CIN = 16, H = 32, HH = 16, L = 3;
    const int HB = 5;                      // log2(H)
    const int N  = in_sizes[0] / CIN;
    const int E  = in_sizes[24] / 2;
    const int NG = out_size;
    const int NP = cdiv(N, 64) * 64;       // padded rows for unguarded WMMA stores
    float* out = (float*)d_out;

    // workspace carve-up (floats); activation buffers padded to NP rows
    float* ws  = (float*)d_ws;
    float* h    = ws;                       // NP*H
    float* Mh   = h   + (long)NP * H;       // NP*H  (messages)
    float* sb   = Mh  + (long)NP * H;       // NP*H  (self-proj)
    float* agg  = sb  + (long)NP * H;       // NP*H
    float* dis  = agg + (long)NP * H;       // N (pad-row reads spill into deg: harmless)
    float* deg  = dis + N;                  // N
    float* hsum = deg + N;                  // NG*H
    unsigned* umax = (unsigned*)(hsum + (long)NG * H);  // NG*H
    float* counts = (float*)(umax + (long)NG * H);      // NG
    float* hg   = counts + NG;              // NG*2H
    float* r1   = hg + (long)NG * 2 * H;    // NG*H   (NG = 128 is a multiple of 64)
    float* r2   = r1 + (long)NG * H;        // NG*HH

    const int B = 256;
    const int GW = cdiv(N, 64);            // WMMA grid (64 rows/block)

    // degree / normalization
    k_zero<<<cdiv(N, B), B, 0, stream>>>(deg, N);
    k_deg_edges<<<cdiv(E, B), B, 0, stream>>>(ei, deg, E);
    k_dis<<<cdiv(N, B), B, 0, stream>>>(deg, dis, N);

    // input projection fused with shared-coef spline activation
    k_wmma_linear<<<GW, 128, 0, stream>>>(x, N, CIN, W_in, H, h, EPI_SPLINE,
                                          b_in, coef_in, blend_in, 0,
                                          nullptr, nullptr, nullptr, nullptr);

    for (int l = 0; l < L; ++l) {
        // message transform fused with per-channel spline + self-loop agg init
        k_wmma_linear<<<GW, 128, 0, stream>>>(h, N, H, Wmsg + l * H * H, H, Mh, EPI_MSG,
                                              nullptr, coef_msg + l * H * NBF,
                                              blend_msg + l * H, 1,
                                              dis, agg, nullptr, nullptr);
        // self projection (raw)
        k_wmma_linear<<<GW, 128, 0, stream>>>(h, N, H, Wself + l * H * H, H, sb, EPI_RAW,
                                              nullptr, nullptr, nullptr, 0,
                                              nullptr, nullptr, nullptr, nullptr);
        // edge scatter-add
        k_edge_scatter<<<cdiv((long)E * 8, B), B, 0, stream>>>(ei, E, dis, Mh, agg);
        // combine + update activation (in place on agg; input from atomics, kept standalone)
        k_spline_add<<<cdiv((long)N * H, B), B, 0, stream>>>(agg, sb, N * H, HB,
                                                             coef_upd + l * H * NBF,
                                                             blend_upd + l * H);
        // output projection fused with layernorm + residual into h
        k_wmma_linear<<<GW, 128, 0, stream>>>(agg, N, H, Wout + l * H * H, H, sb, EPI_LNRES,
                                              nullptr, nullptr, nullptr, 0,
                                              nullptr, h, ln_g + l * H, ln_b + l * H);
    }

    // pooling (mean + max per graph)
    k_zero<<<cdiv((long)NG * H, B), B, 0, stream>>>(hsum, NG * H);
    k_zero<<<cdiv((long)NG * H, B), B, 0, stream>>>((float*)umax, NG * H);
    k_zero<<<cdiv(NG, B), B, 0, stream>>>(counts, NG);
    k_pool_accum<<<cdiv((long)N * H, B), B, 0, stream>>>(h, batch, hsum, umax, counts,
                                                         N * H, HB);
    k_pool_final<<<cdiv(NG * H, B), B, 0, stream>>>(hsum, umax, counts, hg, NG * H, HB);

    // readout MLP, splines fused into the GEMM epilogues
    k_wmma_linear<<<cdiv(NG, 64), 128, 0, stream>>>(hg, NG, 2 * H, W1, H, r1, EPI_SPLINE,
                                                    b1, coef_r1, blend_r1, 1,
                                                    nullptr, nullptr, nullptr, nullptr);
    k_wmma_linear<<<cdiv(NG, 64), 128, 0, stream>>>(r1, NG, H, W2, HH, r2, EPI_SPLINE,
                                                    b2, coef_r2, blend_r2, 1,
                                                    nullptr, nullptr, nullptr, nullptr);
    k_final<<<cdiv(NG, 128), 128, 0, stream>>>(r2, W3, b3, out, NG);
}
